// MultiHeadedAttention_44306882626172
// MI455X (gfx1250) — matrix operand order repaired and checked by reading, not compiled, not run on hardware
//
#include <hip/hip_runtime.h>
#include <hip/hip_bf16.h>

typedef __attribute__((ext_vector_type(16))) _Float16 v16h;
typedef __attribute__((ext_vector_type(8)))  _Float16 v8h;
typedef __attribute__((ext_vector_type(8)))  float    v8f;

// ---------------------------------------------------------------------------
// Problem constants
// ---------------------------------------------------------------------------
#define SEQ_T  2048
#define BATCH  4
#define KDIM   512          // head dim
#define HEADS  8
#define NHEAD  32           // BATCH*HEADS
#define FDIM   4096         // HEADS*KDIM
#define ROWS   8192         // SEQ_T*BATCH

// gfx1250 async global->LDS path (ASYNCcnt), guarded for toolchain portability
#if defined(__has_builtin)
#if __has_builtin(__builtin_amdgcn_global_load_async_to_lds_b128) && \
    __has_builtin(__builtin_amdgcn_s_wait_asynccnt)
#define HAVE_ASYNC_LDS 1
#endif
#endif
#ifndef HAVE_ASYNC_LDS
#define HAVE_ASYNC_LDS 0
#endif

#if HAVE_ASYNC_LDS
// builtin expects: (v4i addrspace(1)* src, v4i addrspace(3)* dst, imm offset, imm cpol)
typedef int i32x4 __attribute__((vector_size(16)));
typedef __attribute__((address_space(1))) i32x4* as1_v4i;
typedef __attribute__((address_space(3))) i32x4* as3_v4i;
#endif

// ---------------------------------------------------------------------------
// WMMA fragment loaders (wave32, 16x16x32 f16)
//
// A operand (16 rows x 32 k, f16): lane L -> row m = L%16, half = L/16.
//   elements 0..7  = K[half*8 + 0..7], elements 8..15 = K[16 + half*8 + 0..7]
// B operand (32 k x 16 cols, f16): lane L -> col n = L%16, half = L/16.
//   elements 0..7  = K[half*8 + 0..7], elements 8..15 = K[16 + half*8 + 0..7]
// C/D (16x16 f32): VGPR j -> row m = j + 8*half, col = lane%16
// ---------------------------------------------------------------------------
__device__ __forceinline__ v16h load_a_frag(const _Float16* base, int row_stride, int lane) {
    const int m    = lane & 15;
    const int half = (lane >> 4) & 1;
    const _Float16* p = base + (size_t)m * row_stride + half * 8;
    v8h lo = *(const v8h*)(p);
    v8h hi = *(const v8h*)(p + 16);
    v16h r;
#pragma unroll
    for (int i = 0; i < 8; ++i) { r[i] = lo[i]; r[i + 8] = hi[i]; }
    return r;
}

__device__ __forceinline__ v16h load_b_frag(const _Float16* base, int row_stride, int lane) {
    const int n    = lane & 15;
    const int half = (lane >> 4) & 1;
    const _Float16* p = base + (size_t)n * row_stride + half * 8;
    v8h lo = *(const v8h*)(p);
    v8h hi = *(const v8h*)(p + 16);
    v16h r;
#pragma unroll
    for (int i = 0; i < 8; ++i) { r[i] = lo[i]; r[i + 8] = hi[i]; }
    return r;
}

__device__ __forceinline__ v8f wmma_f16(v16h a, v16h b, v8f c) {
    return __builtin_amdgcn_wmma_f32_16x16x32_f16(false, a, false, b, (short)0, c, false, false);
}

// ---------------------------------------------------------------------------
// Stage 0: fp32 -> f16 conversions
// ---------------------------------------------------------------------------
__global__ void mha_cvt_f16(const float* __restrict__ src, _Float16* __restrict__ dst, int n) {
    int i = blockIdx.x * 256 + threadIdx.x;
    if (i < n) dst[i] = (_Float16)src[i];
}

// W (KDIM, FDIM) fp32 -> Wt (FDIM, KDIM) f16  (row f contiguous in k)
__global__ void mha_cvt_wT(const float* __restrict__ W, _Float16* __restrict__ Wt) {
    int i = blockIdx.x * 256 + threadIdx.x;               // over FDIM*KDIM
    int f = i >> 9, kk = i & 511;
    Wt[i] = (_Float16)W[(size_t)kk * FDIM + f];
}

// ---------------------------------------------------------------------------
// Stage 1: QKV projection GEMM (ROWS x KDIM) * (KDIM x FDIM), bias + scale.
//   vmode 0: out[t][n][d]  (q, k)      vmode 1: out[n][d][t]  (v transposed)
// grid (ROWS/16, FDIM/64/4), block 128 (4 waves, each a 16x64 tile)
// ---------------------------------------------------------------------------
__global__ void mha_qkv_gemm(const _Float16* __restrict__ xh, const _Float16* __restrict__ wt,
                             const float* __restrict__ bias, float scale,
                             _Float16* __restrict__ out, int vmode) {
    const int lane = threadIdx.x & 31, wid = threadIdx.x >> 5;
    const int half = (lane >> 4) & 1, col = lane & 15;
    const int rt = blockIdx.x;
    const int cbase = (blockIdx.y * 4 + wid) * 64;
    const _Float16* abase = xh + (size_t)rt * 16 * KDIM;

    v8f acc[4] = {};
#pragma unroll
    for (int c = 0; c < 16; ++c) {
        v16h a = load_a_frag(abase + c * 32, KDIM, lane);
#pragma unroll
        for (int nt = 0; nt < 4; ++nt) {
            v16h b = load_b_frag(wt + (size_t)(cbase + nt * 16) * KDIM + c * 32, KDIM, lane);
            acc[nt] = wmma_f16(a, b, acc[nt]);
        }
    }
#pragma unroll
    for (int nt = 0; nt < 4; ++nt) {
        const int f = cbase + nt * 16 + col;
        const float bval = bias[f];
        const int h = f >> 9, d = f & 511;
#pragma unroll
        for (int j = 0; j < 8; ++j) {
            const int r = rt * 16 + j + half * 8;
            const int t = r >> 2, b = r & 3;          // BATCH == 4
            const int nidx = b * HEADS + h;
            const float v = (acc[nt][j] + bval) * scale;
            if (vmode == 0)
                out[((size_t)t * NHEAD + nidx) * KDIM + d] = (_Float16)v;
            else
                out[((size_t)nidx * KDIM + d) * SEQ_T + t] = (_Float16)v;
        }
    }
}

// ---------------------------------------------------------------------------
// Stage 2: flash attention.
// grid (SEQ_T/16, NHEAD), block 128. Each block: one 16-row q tile of one head.
// 4 waves each own a 128-wide slice of the 512 head dim (8 C tiles).
// Per 32-key tile:
//   - all waves cooperatively stage the 32x512 f16 K tile into LDS via the
//     gfx1250 async global->LDS path (ASYNCcnt), barrier,
//   - S = Q*K^T from LDS B-fragments (32 WMMA), online softmax,
//   - P*V from transposed-V global B-fragments (8 WMMA).
// ---------------------------------------------------------------------------
__global__ void mha_flash_attn(const _Float16* __restrict__ q, const _Float16* __restrict__ k,
                               const _Float16* __restrict__ vT, _Float16* __restrict__ ctx) {
    const int lane = threadIdx.x & 31, wid = threadIdx.x >> 5;
    const int half = (lane >> 4) & 1, col = lane & 15;
    const int qt = blockIdx.x, n = blockIdx.y;
    const int dbase = wid * 128;
    const int rowstride = NHEAD * KDIM;

    __shared__ _Float16 kstage[32 * KDIM];       // 32 keys x 512 dims = 32 KB
    __shared__ _Float16 pstage[4][16 * 32];      // per-wave P staging (C->A relayout)
    _Float16* pst = &pstage[wid][0];

    // Q fragments resident in registers: 16 chunks of k (16 x 512 f16)
    const _Float16* qbase = q + ((size_t)qt * 16 * NHEAD + n) * KDIM;
    v16h qf[16];
#pragma unroll
    for (int c = 0; c < 16; ++c) qf[c] = load_a_frag(qbase + c * 32, rowstride, lane);

    v8f acc[8] = {};
    float mrun[8], lrun[8];
#pragma unroll
    for (int j = 0; j < 8; ++j) { mrun[j] = -1e30f; lrun[j] = 0.0f; }

    for (int kt = 0; kt < SEQ_T / 32; ++kt) {
        // ---- cooperative K-tile stage: 32 keys x 1KB rows = 2048 x 16B chunks
        const _Float16* kglob = k + ((size_t)(kt * 32) * NHEAD + n) * KDIM;
#pragma unroll
        for (int it = 0; it < 16; ++it) {
            const int c16 = threadIdx.x + it * 128;          // 0..2047
            const int key = c16 >> 6, off = (c16 & 63) * 8;  // 8 f16 = 16B
            const _Float16* src = kglob + (size_t)key * rowstride + off;
            _Float16* dst = &kstage[key * KDIM + off];
#if HAVE_ASYNC_LDS
            __builtin_amdgcn_global_load_async_to_lds_b128((as1_v4i)src, (as3_v4i)dst, 0, 0);
#else
            *(v8h*)dst = *(const v8h*)src;
#endif
        }
#if HAVE_ASYNC_LDS
        __builtin_amdgcn_s_wait_asynccnt(0);
#endif
        __syncthreads();

        if (kt + 1 < SEQ_T / 32)
            __builtin_prefetch(kglob + (size_t)32 * rowstride, 0, 1);   // global_prefetch_b8

        // ---- S = Q * K^T from LDS (keys 0-15 -> s0, keys 16-31 -> s1)
        v8f s0 = {}, s1 = {};
#pragma unroll
        for (int c = 0; c < 16; ++c) {
            v16h b0 = load_b_frag(&kstage[0]          + c * 32, KDIM, lane);
            v16h b1 = load_b_frag(&kstage[16 * KDIM]  + c * 32, KDIM, lane);
            s0 = wmma_f16(qf[c], b0, s0);
            s1 = wmma_f16(qf[c], b1, s1);
        }

        // ---- online softmax over this 32-key tile (row = j + 8*half) ----
#pragma unroll
        for (int j = 0; j < 8; ++j) {
            float mx = fmaxf(s0[j], s1[j]);
            mx = fmaxf(mx, __shfl_xor(mx, 1, 32));
            mx = fmaxf(mx, __shfl_xor(mx, 2, 32));
            mx = fmaxf(mx, __shfl_xor(mx, 4, 32));
            mx = fmaxf(mx, __shfl_xor(mx, 8, 32));
            const float mnew = fmaxf(mrun[j], mx);
            const float corr = __expf(mrun[j] - mnew);
            const float p0 = __expf(s0[j] - mnew);
            const float p1 = __expf(s1[j] - mnew);
            float rs = p0 + p1;
            rs += __shfl_xor(rs, 1, 32);
            rs += __shfl_xor(rs, 2, 32);
            rs += __shfl_xor(rs, 4, 32);
            rs += __shfl_xor(rs, 8, 32);
            lrun[j] = lrun[j] * corr + rs;
            mrun[j] = mnew;
            const int row = j + half * 8;
            pst[row * 32 + col]      = (_Float16)p0;
            pst[row * 32 + col + 16] = (_Float16)p1;
#pragma unroll
            for (int dt = 0; dt < 8; ++dt) acc[dt][j] *= corr;
        }

        // ---- P * V over this wave's d-slice ----
        v16h pa = load_a_frag(pst, 32, lane);
#pragma unroll
        for (int dt = 0; dt < 8; ++dt) {
            const _Float16* vb = vT + ((size_t)n * KDIM + dbase + dt * 16) * SEQ_T + kt * 32;
            v16h b = load_b_frag(vb, SEQ_T, lane);
            acc[dt] = wmma_f16(pa, b, acc[dt]);
        }

        __syncthreads();   // all waves done with kstage before next overwrite
    }

    // normalize and write ctx[t][n][d] as f16
#pragma unroll
    for (int dt = 0; dt < 8; ++dt) {
#pragma unroll
        for (int j = 0; j < 8; ++j) {
            const int t = qt * 16 + j + half * 8;
            const int d = dbase + dt * 16 + col;
            ctx[((size_t)t * NHEAD + n) * KDIM + d] = (_Float16)(acc[dt][j] / lrun[j]);
        }
    }
}

// ---------------------------------------------------------------------------
// Stage 3: output projection ctx(ROWS, FDIM) * Wo(FDIM, 8) + bo
// grid ROWS, block 256
// ---------------------------------------------------------------------------
__global__ void mha_out_proj(const _Float16* __restrict__ ctx, const float* __restrict__ Wo,
                             const float* __restrict__ bo, float* __restrict__ out) {
    const int rb = blockIdx.x;               // t*BATCH + b
    const int t = rb >> 2, b = rb & 3;
    const int tid = threadIdx.x;
    float a[8] = {};
    for (int f = tid; f < FDIM; f += 256) {
        const int h = f >> 9, d = f & 511;
        const float c = (float)ctx[((size_t)t * NHEAD + b * HEADS + h) * KDIM + d];
#pragma unroll
        for (int ho = 0; ho < 8; ++ho) a[ho] += c * Wo[(size_t)f * 8 + ho];
    }
    __shared__ float red[256][8];
#pragma unroll
    for (int ho = 0; ho < 8; ++ho) red[tid][ho] = a[ho];
    __syncthreads();
    for (int s = 128; s > 0; s >>= 1) {
        if (tid < s)
#pragma unroll
            for (int ho = 0; ho < 8; ++ho) red[tid][ho] += red[tid + s][ho];
        __syncthreads();
    }
    if (tid < 8) out[(size_t)rb * 8 + tid] = red[0][tid] + bo[tid];
}

// ---------------------------------------------------------------------------
// Launch
// ---------------------------------------------------------------------------
extern "C" void kernel_launch(void* const* d_in, const int* in_sizes, int n_in,
                              void* d_out, int out_size, void* d_ws, size_t ws_size,
                              hipStream_t stream) {
    (void)in_sizes; (void)n_in; (void)out_size; (void)ws_size;
    const float* x  = (const float*)d_in[0];
    const float* Wq = (const float*)d_in[1];
    const float* bq = (const float*)d_in[2];
    const float* Wk = (const float*)d_in[3];
    const float* bk = (const float*)d_in[4];
    const float* Wv = (const float*)d_in[5];
    const float* bv = (const float*)d_in[6];
    const float* Wo = (const float*)d_in[7];
    const float* bo = (const float*)d_in[8];
    float* out = (float*)d_out;

    char* ws = (char*)d_ws;
    size_t off = 0;
    auto take = [&](size_t bytes) { char* p = ws + off; off += (bytes + 255) & ~(size_t)255; return p; };

    _Float16* xh  = (_Float16*)take((size_t)ROWS * KDIM * 2);
    _Float16* wtq = (_Float16*)take((size_t)FDIM * KDIM * 2);
    _Float16* wtk = (_Float16*)take((size_t)FDIM * KDIM * 2);
    _Float16* wtv = (_Float16*)take((size_t)FDIM * KDIM * 2);
    _Float16* qws = (_Float16*)take((size_t)SEQ_T * NHEAD * KDIM * 2);
    _Float16* kws = (_Float16*)take((size_t)SEQ_T * NHEAD * KDIM * 2);
    _Float16* vtw = (_Float16*)take((size_t)SEQ_T * NHEAD * KDIM * 2);
    _Float16* ctx = (_Float16*)take((size_t)SEQ_T * NHEAD * KDIM * 2);

    const float scale = 0.21022410f;   // 512^{-1/4}

    const int nx = ROWS * KDIM;        // 4,194,304
    mha_cvt_f16<<<nx / 256, 256, 0, stream>>>(x, xh, nx);
    const int nw = FDIM * KDIM;        // 2,097,152
    mha_cvt_wT<<<nw / 256, 256, 0, stream>>>(Wq, wtq);
    mha_cvt_wT<<<nw / 256, 256, 0, stream>>>(Wk, wtk);
    mha_cvt_wT<<<nw / 256, 256, 0, stream>>>(Wv, wtv);

    dim3 ggrid(ROWS / 16, FDIM / 64 / 4);   // (512, 16)
    mha_qkv_gemm<<<ggrid, 128, 0, stream>>>(xh, wtq, bq, scale, qws, 0);
    mha_qkv_gemm<<<ggrid, 128, 0, stream>>>(xh, wtk, bk, scale, kws, 0);
    mha_qkv_gemm<<<ggrid, 128, 0, stream>>>(xh, wtv, bv, 1.0f, vtw, 1);

    dim3 fgrid(SEQ_T / 16, NHEAD);          // (128, 32)
    mha_flash_attn<<<fgrid, 128, 0, stream>>>(qws, kws, vtw, ctx);

    mha_out_proj<<<ROWS, 256, 0, stream>>>(ctx, Wo, bo, out);
}
